// ClassDiagramGNN_77678778515729
// MI455X (gfx1250) — compile-verified, matrix operand order checked by reading, not run on hardware
//
#include <hip/hip_runtime.h>
#include <math.h>

// ---- CDNA5 WMMA vector types (wave32) ----
typedef __attribute__((ext_vector_type(16))) __bf16 v16bf;
typedef __attribute__((ext_vector_type(8)))  __bf16 v8bf;
typedef __attribute__((ext_vector_type(8)))  float  v8f;

#define N_NODES   50000
#define F_INDIM   512
#define HIDC      128
#define HEADS1    4
#define N_EDGES   800000
#define E_TOT     (N_EDGES + N_NODES)
#define NEG_SLOPE 0.2f

// float -> bf16 round-to-nearest-even
__device__ __forceinline__ unsigned short f2bf(float f) {
  unsigned u = __float_as_uint(f);
  u += 0x7FFFu + ((u >> 16) & 1u);
  return (unsigned short)(u >> 16);
}

__device__ __forceinline__ void atomic_max_f32(float* addr, float val) {
  // int-max for non-negative, uint-min for negative: total order matches float
  if (val >= 0.0f) atomicMax((int*)addr, __float_as_int(val));
  else             atomicMin((unsigned int*)addr, __float_as_uint(val));
}

// ---------------- elementwise helpers ----------------
__global__ void cvt_bf16_kernel(const float* __restrict__ src,
                                unsigned short* __restrict__ dst, long n) {
  long i = blockIdx.x * (long)blockDim.x + threadIdx.x;
  long st = (long)gridDim.x * blockDim.x;
  for (; i < n; i += st) dst[i] = f2bf(src[i]);
}

__global__ void fill_kernel(float* __restrict__ p, float v, long n) {
  long i = blockIdx.x * (long)blockDim.x + threadIdx.x;
  long st = (long)gridDim.x * blockDim.x;
  for (; i < n; i += st) p[i] = v;
}

// Pack row-major W[K,Nout] (f32) into WMMA-B-operand tile layout (bf16):
// tile(kt,ct) is 32x16; packed[tile*512 + lane*16 + j]:
//   n = ct*16 + lane%16, khi = lane/16,
//   j<8  -> k = kt*32 +      khi*8 + j
//   j>=8 -> k = kt*32 + 16 + khi*8 + (j-8)
__global__ void pack_w_kernel(const float* __restrict__ W,
                              unsigned short* __restrict__ Wp,
                              int K, int Nout) {
  int ntiles = Nout >> 4;
  long total = (long)K * Nout;
  long i = blockIdx.x * (long)blockDim.x + threadIdx.x;
  long st = (long)gridDim.x * blockDim.x;
  for (; i < total; i += st) {
    int tile = (int)(i >> 9);
    int r    = (int)(i & 511);
    int l    = r >> 4;
    int j    = r & 15;
    int kt   = tile / ntiles;
    int ct   = tile - kt * ntiles;
    int khi  = l >> 4;
    int n    = ct * 16 + (l & 15);
    int k    = kt * 32 + khi * 8 + ((j < 8) ? j : (16 + j - 8));
    Wp[i] = f2bf(W[(size_t)k * Nout + n]);
  }
}

// ------- bf16 WMMA GEMM, register-blocked 16x64 per wave -------
// C[M,Nout] = A[M,K] * B[K,Nout]. A row-major bf16; Bp pre-packed tiles.
// Wave rt = blockIdx.x*4+wave computes rows [rt*16, rt*16+16) and the
// 64-column strip blockIdx.y*64: 4 accumulators share one A fragment,
// so each k-step is 2 A-loads + 4 B-loads feeding 4 WMMAs.
__global__ void gemm_bf16_wmma_kernel(const unsigned short* __restrict__ A,
                                      const unsigned short* __restrict__ Bp,
                                      float* __restrict__ C,
                                      int Nout, int K, int mtiles) {
  int lane = threadIdx.x & 31;
  int wave = threadIdx.x >> 5;
  int rt   = blockIdx.x * 4 + wave;
  if (rt >= mtiles) return;
  int row0 = rt * 16;
  int ct0  = blockIdx.y * 4;          // first of 4 column tiles
  int ntiles = Nout >> 4;
  int ktiles = K >> 5;
  int m   = lane & 15;
  int khi = lane >> 4;

  const unsigned short* pA = A + (size_t)(row0 + m) * K + khi * 8;
  const unsigned short* pB = Bp + (size_t)ct0 * 512 + (size_t)lane * 16;
  size_t bstride = (size_t)ntiles * 512;

  v8f acc[4];
  acc[0] = (v8f){}; acc[1] = (v8f){}; acc[2] = (v8f){}; acc[3] = (v8f){};

  for (int kt = 0; kt < ktiles; ++kt) {
    v8bf alo = *(const v8bf*)(pA + (size_t)kt * 32);
    v8bf ahi = *(const v8bf*)(pA + (size_t)kt * 32 + 16);
    v16bf a = __builtin_shufflevector(alo, ahi,
        0,1,2,3,4,5,6,7,8,9,10,11,12,13,14,15);
    const unsigned short* pbk = pB + (size_t)kt * bstride;
    v16bf b0 = *(const v16bf*)(pbk);
    v16bf b1 = *(const v16bf*)(pbk + 512);
    v16bf b2 = *(const v16bf*)(pbk + 1024);
    v16bf b3 = *(const v16bf*)(pbk + 1536);
    acc[0] = __builtin_amdgcn_wmma_f32_16x16x32_bf16(
        false, a, false, b0, (short)0, acc[0], false, false);
    acc[1] = __builtin_amdgcn_wmma_f32_16x16x32_bf16(
        false, a, false, b1, (short)0, acc[1], false, false);
    acc[2] = __builtin_amdgcn_wmma_f32_16x16x32_bf16(
        false, a, false, b2, (short)0, acc[2], false, false);
    acc[3] = __builtin_amdgcn_wmma_f32_16x16x32_bf16(
        false, a, false, b3, (short)0, acc[3], false, false);
  }

  // C layout: VGPR v -> row (v + khi*8), col = lane%16
  int rbase = row0 + khi * 8;
  int coll  = lane & 15;
#pragma unroll
  for (int t = 0; t < 4; ++t) {
    int col = (ct0 + t) * 16 + coll;
#pragma unroll
    for (int v = 0; v < 8; ++v)
      C[(size_t)(rbase + v) * Nout + col] = acc[t][v];
  }
}

// ---------------- attention score dot-products ----------------
// one wave per (node, head); C = 128 channels = 32 lanes * float4
__global__ void scores_kernel(const float* __restrict__ h,
                              const float* __restrict__ a_src,
                              const float* __restrict__ a_dst,
                              float* __restrict__ s_src,
                              float* __restrict__ s_dst,
                              int H, int n_nodes) {
  int gwave = (blockIdx.x * blockDim.x + threadIdx.x) >> 5;
  int lane  = threadIdx.x & 31;
  if (gwave >= n_nodes * H) return;
  int node = gwave / H;
  int head = gwave - node * H;
  const float4* hv = (const float4*)(h + ((size_t)node * H + head) * HIDC);
  const float4* as = (const float4*)(a_src + (size_t)head * HIDC);
  const float4* ad = (const float4*)(a_dst + (size_t)head * HIDC);
  float4 x = hv[lane];
  float4 s = as[lane];
  float4 d = ad[lane];
  float ps = x.x*s.x + x.y*s.y + x.z*s.z + x.w*s.w;
  float pd = x.x*d.x + x.y*d.y + x.z*d.z + x.w*d.w;
#pragma unroll
  for (int off = 16; off > 0; off >>= 1) {
    ps += __shfl_xor(ps, off, 32);
    pd += __shfl_xor(pd, off, 32);
  }
  if (lane == 0) {
    s_src[(size_t)node * H + head] = ps;
    s_dst[(size_t)node * H + head] = pd;
  }
}

__device__ __forceinline__ void edge_sd(const int* ei, int i, int& s, int& d) {
  if (i < N_EDGES) { s = ei[i]; d = ei[N_EDGES + i]; }
  else             { s = d = i - N_EDGES; }            // self loops
}

__global__ void edge_max_kernel(const int* __restrict__ ei,
                                const float* __restrict__ ssrc,
                                const float* __restrict__ sdst,
                                float* __restrict__ m, int H) {
  int i = blockIdx.x * blockDim.x + threadIdx.x;
  if (i >= E_TOT) return;
  int s, d; edge_sd(ei, i, s, d);
  for (int h = 0; h < H; ++h) {
    float e = ssrc[(size_t)s * H + h] + sdst[(size_t)d * H + h];
    e = (e > 0.f) ? e : NEG_SLOPE * e;
    atomic_max_f32(&m[(size_t)d * H + h], e);
  }
}

__global__ void edge_sum_kernel(const int* __restrict__ ei,
                                const float* __restrict__ ssrc,
                                const float* __restrict__ sdst,
                                const float* __restrict__ m,
                                float* __restrict__ denom, int H) {
  int i = blockIdx.x * blockDim.x + threadIdx.x;
  if (i >= E_TOT) return;
  int s, d; edge_sd(ei, i, s, d);
  for (int h = 0; h < H; ++h) {
    float e = ssrc[(size_t)s * H + h] + sdst[(size_t)d * H + h];
    e = (e > 0.f) ? e : NEG_SLOPE * e;
    atomicAdd(&denom[(size_t)d * H + h], __expf(e - m[(size_t)d * H + h]));
  }
}

// one wave per (edge, head): gather float4 of h[src], scale by alpha, scatter-add
__global__ void edge_agg_kernel(const int* __restrict__ ei,
                                const float* __restrict__ h,
                                const float* __restrict__ ssrc,
                                const float* __restrict__ sdst,
                                const float* __restrict__ m,
                                const float* __restrict__ denom,
                                float* __restrict__ agg, int H) {
  int gwave = (blockIdx.x * blockDim.x + threadIdx.x) >> 5;
  int lane  = threadIdx.x & 31;
  if (gwave >= E_TOT * H) return;
  int e_id = gwave / H;
  int head = gwave - e_id * H;
  int s, d; edge_sd(ei, e_id, s, d);
  float e = ssrc[(size_t)s * H + head] + sdst[(size_t)d * H + head];
  e = (e > 0.f) ? e : NEG_SLOPE * e;
  float p = __expf(e - m[(size_t)d * H + head]);
  float alpha = p / (denom[(size_t)d * H + head] + 1e-16f);
  const float4* hv = (const float4*)(h + ((size_t)s * H + head) * HIDC);
  float4 x = hv[lane];
  float* o = agg + ((size_t)d * H + head) * HIDC + lane * 4;
  atomicAdd(o + 0, x.x * alpha);
  atomicAdd(o + 1, x.y * alpha);
  atomicAdd(o + 2, x.z * alpha);
  atomicAdd(o + 3, x.w * alpha);
}

__global__ void bias_elu_bf16_kernel(const float* __restrict__ agg,
                                     const float* __restrict__ b,
                                     unsigned short* __restrict__ out,
                                     long n, int F) {
  long i = blockIdx.x * (long)blockDim.x + threadIdx.x;
  long st = (long)gridDim.x * blockDim.x;
  for (; i < n; i += st) {
    float v = agg[i] + b[i % F];
    v = (v > 0.f) ? v : (__expf(v) - 1.f);   // ELU(alpha=1)
    out[i] = f2bf(v);
  }
}

__global__ void bias_add_kernel(const float* __restrict__ agg,
                                const float* __restrict__ b,
                                float* __restrict__ out, long n, int F) {
  long i = blockIdx.x * (long)blockDim.x + threadIdx.x;
  long st = (long)gridDim.x * blockDim.x;
  for (; i < n; i += st) out[i] = agg[i] + b[i % F];
}

extern "C" void kernel_launch(void* const* d_in, const int* in_sizes, int n_in,
                              void* d_out, int out_size, void* d_ws, size_t ws_size,
                              hipStream_t stream) {
  (void)in_sizes; (void)n_in; (void)out_size; (void)ws_size;
  const float* x      = (const float*)d_in[0];
  const int*   ei     = (const int*)  d_in[1];
  const float* W1     = (const float*)d_in[2];
  const float* a_src1 = (const float*)d_in[3];
  const float* a_dst1 = (const float*)d_in[4];
  const float* b1     = (const float*)d_in[5];
  const float* W2     = (const float*)d_in[6];
  const float* a_src2 = (const float*)d_in[7];
  const float* a_dst2 = (const float*)d_in[8];
  const float* b2     = (const float*)d_in[9];
  float* out = (float*)d_out;

  char* ws = (char*)d_ws;
  size_t off = 0;
  auto alloc = [&](size_t bytes) -> void* {
    void* p = ws + off;
    off = (off + bytes + 255) & ~(size_t)255;
    return p;
  };
  unsigned short* xbf = (unsigned short*)alloc((size_t)N_NODES * 512 * 2);
  unsigned short* W1p = (unsigned short*)alloc((size_t)512 * 512 * 2);
  unsigned short* W2p = (unsigned short*)alloc((size_t)512 * 128 * 2);
  float* h1   = (float*)alloc((size_t)N_NODES * 512 * 4);
  float* agg  = (float*)alloc((size_t)N_NODES * 512 * 4);
  float* ssrc = (float*)alloc((size_t)N_NODES * HEADS1 * 4);
  float* sdst = (float*)alloc((size_t)N_NODES * HEADS1 * 4);
  float* mbuf = (float*)alloc((size_t)N_NODES * HEADS1 * 4);
  float* dbuf = (float*)alloc((size_t)N_NODES * HEADS1 * 4);
  unsigned short* h1act = xbf;   // reuse: x(bf16) dead after GEMM1
  float* h2   = h1;              // reuse: h1 dead after layer-1 aggregation
  float* agg2 = agg;             // reuse: agg dead after ELU pass

  const int mtiles = N_NODES / 16;           // 3125 (exact)
  const int gemm_bx = (mtiles + 3) / 4;      // 4 row-tiles per block (4 waves)

  // ---- layer 1 ----
  cvt_bf16_kernel<<<2048, 256, 0, stream>>>(x, xbf, (long)N_NODES * 512);
  pack_w_kernel<<<1024, 256, 0, stream>>>(W1, W1p, 512, 512);
  pack_w_kernel<<<256, 256, 0, stream>>>(W2, W2p, 512, 128);

  gemm_bf16_wmma_kernel<<<dim3(gemm_bx, 512 / 64), 128, 0, stream>>>(
      xbf, W1p, h1, 512, 512, mtiles);

  scores_kernel<<<(N_NODES * HEADS1 + 3) / 4, 128, 0, stream>>>(
      h1, a_src1, a_dst1, ssrc, sdst, HEADS1, N_NODES);

  fill_kernel<<<512, 256, 0, stream>>>(mbuf, -INFINITY, (long)N_NODES * HEADS1);
  fill_kernel<<<512, 256, 0, stream>>>(dbuf, 0.f, (long)N_NODES * HEADS1);
  fill_kernel<<<4096, 256, 0, stream>>>(agg, 0.f, (long)N_NODES * 512);

  edge_max_kernel<<<(E_TOT + 255) / 256, 256, 0, stream>>>(ei, ssrc, sdst, mbuf, HEADS1);
  edge_sum_kernel<<<(E_TOT + 255) / 256, 256, 0, stream>>>(ei, ssrc, sdst, mbuf, dbuf, HEADS1);
  edge_agg_kernel<<<(E_TOT * HEADS1 + 3) / 4, 128, 0, stream>>>(
      ei, h1, ssrc, sdst, mbuf, dbuf, agg, HEADS1);

  bias_elu_bf16_kernel<<<2048, 256, 0, stream>>>(agg, b1, h1act, (long)N_NODES * 512, 512);

  // ---- layer 2 ----
  gemm_bf16_wmma_kernel<<<dim3(gemm_bx, 128 / 64), 128, 0, stream>>>(
      h1act, W2p, h2, 128, 512, mtiles);

  scores_kernel<<<(N_NODES + 3) / 4, 128, 0, stream>>>(
      h2, a_src2, a_dst2, ssrc, sdst, 1, N_NODES);

  fill_kernel<<<256, 256, 0, stream>>>(mbuf, -INFINITY, (long)N_NODES);
  fill_kernel<<<256, 256, 0, stream>>>(dbuf, 0.f, (long)N_NODES);
  fill_kernel<<<2048, 256, 0, stream>>>(agg2, 0.f, (long)N_NODES * 128);

  edge_max_kernel<<<(E_TOT + 255) / 256, 256, 0, stream>>>(ei, ssrc, sdst, mbuf, 1);
  edge_sum_kernel<<<(E_TOT + 255) / 256, 256, 0, stream>>>(ei, ssrc, sdst, mbuf, dbuf, 1);
  edge_agg_kernel<<<(E_TOT + 3) / 4, 128, 0, stream>>>(
      ei, h2, ssrc, sdst, mbuf, dbuf, agg2, 1);

  bias_add_kernel<<<1024, 256, 0, stream>>>(agg2, b2, out, (long)N_NODES * 128, 128);
}
